// GeniePathLayer_56341380989602
// MI455X (gfx1250) — matrix operand order repaired and checked by reading, not compiled, not run on hardware
//
#include <hip/hip_runtime.h>
#include <hip/hip_bf16.h>

// ---------------------------------------------------------------------------
// GeniePath layer for MI455X (gfx1250, wave32):
//   bf16 WMMA GEMMs, all LDS staging via Tensor Data Mover, double-buffered.
// ---------------------------------------------------------------------------

typedef __attribute__((ext_vector_type(16))) __bf16 v16bf;
typedef __attribute__((ext_vector_type(8)))  __bf16 v8bf;
typedef __attribute__((ext_vector_type(4)))  __bf16 v4bf;
typedef __attribute__((ext_vector_type(8)))  float  v8f;
typedef unsigned int u32;
typedef __attribute__((ext_vector_type(4)))  u32    u32x4;
typedef __attribute__((ext_vector_type(8)))  int    i32x8;
typedef __attribute__((ext_vector_type(4)))  int    i32x4;

#define DIM      256
#define TM       128
#define TN       128
#define TK        32
#define LDAS      40   // LDS row stride (32 + 8 bf16 pad, produced by TDM pad)
#define LDBS      40   // LDS col stride (32 + 8 bf16 pad, produced by TDM pad)
#define NKSLAB     8   // 256 / 32 k-slabs per phase
#define NEG_SLOPE 0.2f

#define HAS_TDM __has_builtin(__builtin_amdgcn_tensor_load_to_lds)

// ------------------------------ helpers ------------------------------------

__device__ __forceinline__ float atomicMaxFloat(float* addr, float value) {
    if (value >= 0.f)
        return __int_as_float(atomicMax((int*)addr, __float_as_int(value)));
    else
        return __uint_as_float(atomicMin((unsigned int*)addr, __float_as_uint(value)));
}

__device__ __forceinline__ float sigmoidf(float x) {
    return 1.0f / (1.0f + __expf(-x));
}

__device__ __forceinline__ v16bf concat8(v8bf a, v8bf b) {
    return __builtin_shufflevector(a, b, 0, 1, 2, 3, 4, 5, 6, 7,
                                   8, 9, 10, 11, 12, 13, 14, 15);
}

// Tensor Data Mover 2-D copy: tile (tile0 x tile1) of bf16, global row stride
// stride0 (elements), into LDS with 4 DWORDs (8 bf16) of padding inserted
// every 16 DWORDs (32 bf16) -> LDS stride 40 per 32-element line.
// Rows/cols beyond tensor_dim are zero-filled by the TDM (handles M tail).
// D# per CDNA5 ISA 8.3/8.4.
__device__ __forceinline__ void tdm_copy_2d(const __bf16* gsrc, void* lds_dst,
                                            u32 tile0, u32 tile1,
                                            u32 tdim0, u32 tdim1, u32 stride0) {
#if HAS_TDM
    unsigned long long ga = (unsigned long long)gsrc;
    u32 lds = (u32)(unsigned long long)lds_dst;   // low 32b = LDS byte offset
    u32x4 g0;
    g0[0] = 1u;                                              // count = 1
    g0[1] = lds;                                             // lds_addr
    g0[2] = (u32)ga;                                         // global_addr lo
    g0[3] = (u32)((ga >> 32) & 0x01FFFFFFu) | (2u << 30);    // addr hi | type=2
    i32x8 g1;
    g1[0] = (int)((1u << 16) | (1u << 20) | (3u << 22) | (3u << 25));
            // data_size=2B | pad_enable | pad_interval=16dw | pad_amount=4dw
    g1[1] = (int)((tdim0 & 0xFFFFu) << 16);                  // tensor_dim0 lo
    g1[2] = (int)((tdim0 >> 16) | ((tdim1 & 0xFFFFu) << 16));// dim0 hi|dim1 lo
    g1[3] = (int)((tdim1 >> 16) | (tile0 << 16));            // dim1 hi|tile0
    g1[4] = (int)(tile1 & 0xFFFFu);                          // tile1 (tile2=0)
    g1[5] = (int)stride0;                                    // dim0_stride lo
    g1[6] = 0;                                               // stride hi = 0
    g1[7] = 0;
    i32x4 z4 = {0, 0, 0, 0};
#if defined(__clang_major__) && (__clang_major__ >= 23)
    i32x8 z8 = {0, 0, 0, 0, 0, 0, 0, 0};
    __builtin_amdgcn_tensor_load_to_lds(g0, g1, z4, z4, z8, 0);
#else
    __builtin_amdgcn_tensor_load_to_lds(g0, g1, z4, z4, 0);
#endif
#else
    (void)gsrc; (void)lds_dst; (void)tile0; (void)tile1;
    (void)tdim0; (void)tdim1; (void)stride0;
#endif
}

// ------------------------- conversion kernels -------------------------------

// W_gat [256x256] and W_ih/W_hh [1024x256] -> bf16, B-swizzled layout:
//   B_sw[(k/32)*C + n][kk], kk = k%32  (C = output columns)
// -> a 128-col x 32-K tile is one contiguous 8KB slab; a lane's 16-element
//    WMMA B-fragment is two 16B runs, at k = 8h and k = 16 + 8h (h = lane / 16).
__global__ __launch_bounds__(256)
void k_cvt_weights(const float* __restrict__ Wg, const float* __restrict__ Wih,
                   const float* __restrict__ Whh, __bf16* __restrict__ Wg_sw,
                   __bf16* __restrict__ WihT_sw, __bf16* __restrict__ WhhT_sw) {
    int i = blockIdx.x * 256 + threadIdx.x;
    if (i < DIM * DIM) {   // Wg: B[k][n] = Wg[k*256+n]
        int k = i >> 8, n = i & (DIM - 1);
        Wg_sw[((size_t)(k >> 5) * DIM + n) * TK + (k & 31)] = (__bf16)Wg[i];
    }
    if (i < 4 * DIM * DIM) {   // W_ih/W_hh: B[k][j] = W[j*256+k]
        int j = i >> 8, k = i & (DIM - 1);
        size_t o = ((size_t)(k >> 5) * (4 * DIM) + j) * TK + (k & 31);
        WihT_sw[o] = (__bf16)Wih[i];
        WhhT_sw[o] = (__bf16)Whh[i];
    }
}

__global__ __launch_bounds__(256)
void k_cvt_f32_to_bf16_v4(const float* __restrict__ src,
                          __bf16* __restrict__ dst, int total4) {
    int i = blockIdx.x * 256 + threadIdx.x;
    if (i >= total4) return;
    float4 v = ((const float4*)src)[i];
    v4bf o;
    o[0] = (__bf16)v.x; o[1] = (__bf16)v.y;
    o[2] = (__bf16)v.z; o[3] = (__bf16)v.w;
    ((v4bf*)dst)[i] = o;
}

// init: m = -inf, denom = 0, out = 0
__global__ __launch_bounds__(256)
void k_init(float* __restrict__ m, float* __restrict__ denom,
            float* __restrict__ out, int N) {
    int i = blockIdx.x * 256 + threadIdx.x;
    if (i < N) { m[i] = -__builtin_inff(); denom[i] = 0.f; }
    if (i < N * DIM) out[i] = 0.f;
}

// --------------------- unified bf16 WMMA GEMM -------------------------------
// C[N x ldc] = sum_ph A_ph[N x 256](bf16 row-major) @ B_ph[256 x ldc](bf16 swz)
// nslab = 8 * phases.  Block tile 128x128, wave tile 32x64, double-buffered
// LDS fed exclusively by the Tensor Data Mover.
__global__ __launch_bounds__(256)
void k_gemm(const __bf16* __restrict__ A1, const __bf16* __restrict__ A2,
            const __bf16* __restrict__ B1, const __bf16* __restrict__ B2,
            float* __restrict__ C, int N, int ldc, int nslab) {
    __shared__ __align__(16) __bf16 As[2][TM * LDAS];
    __shared__ __align__(16) __bf16 Bs[2][TN * LDBS];
    const int tid  = threadIdx.x;
    const int lane = tid & 31;
    const int wave = tid >> 5;       // 8 waves
    const int wr   = wave >> 1;      // 0..3: 32-row band
    const int wc   = wave & 1;       // 0..1: 64-col band
    const int row0 = blockIdx.x * TM;
    const int col0 = blockIdx.y * TN;
    const u32 rows_left = (u32)(N - row0);   // TDM zero-fills rows beyond this

#if HAS_TDM
    // ---- TDM issue for slab s into buffer b (wave 0 only) ----
    auto issue = [&](int s, int b) {
        const __bf16* A = (s < NKSLAB) ? A1 : A2;
        const __bf16* B = (s < NKSLAB) ? B1 : B2;
        const int kb = (s & (NKSLAB - 1)) * TK;
        // A: 2-D tile 32 x 128, global row stride DIM
        tdm_copy_2d(A + (size_t)row0 * DIM + kb, (void*)&As[b][0],
                    TK, TM, TK, rows_left, DIM);
        // B: contiguous 128 cols x 32 = 8KB slab
        tdm_copy_2d(B + ((size_t)(kb >> 5) * ldc + col0) * TK, (void*)&Bs[b][0],
                    TN * TK, 1, TN * TK, 1, TN * TK);
    };
    if (tid < 32) {
        issue(0, 0);
        __builtin_amdgcn_s_wait_tensorcnt(0);
    }
    __syncthreads();
#endif

    v8f acc[2][4] = {};
    for (int i = 0; i < nslab; ++i) {
        const int buf = i & 1;
#if HAS_TDM
        if (tid < 32 && (i + 1) < nslab) issue(i + 1, buf ^ 1);
#else
        {   // fallback: manual staging of slab i into buf
            const __bf16* A = (i < NKSLAB) ? A1 : A2;
            const __bf16* B = (i < NKSLAB) ? B1 : B2;
            const int kb = (i & (NKSLAB - 1)) * TK;
            int r = tid >> 2, off = (tid & 3) * 8;
            int g0 = row0 + r;       g0 = g0 < N ? g0 : N - 1;
            int g1 = row0 + r + 64;  g1 = g1 < N ? g1 : N - 1;
            v8bf a0 = *(const v8bf*)&A[(size_t)g0 * DIM + kb + off];
            v8bf a1 = *(const v8bf*)&A[(size_t)g1 * DIM + kb + off];
            const __bf16* bt = B + ((size_t)(kb >> 5) * ldc + col0) * TK;
            v8bf b0 = *(const v8bf*)&bt[(size_t)tid * 8];
            v8bf b1 = *(const v8bf*)&bt[(size_t)(tid + 256) * 8];
            *(v8bf*)&As[buf][r * LDAS + off] = a0;
            *(v8bf*)&As[buf][(r + 64) * LDAS + off] = a1;
            int c0i = tid >> 2,         o0 = (tid & 3) * 8;
            int c1i = (tid + 256) >> 2, o1 = (tid & 3) * 8;
            *(v8bf*)&Bs[buf][c0i * LDBS + o0] = b0;
            *(v8bf*)&Bs[buf][c1i * LDBS + o1] = b1;
            __syncthreads();
        }
#endif
        // ---- fragments: contiguous 16B LDS reads ----
        v16bf af[2], bfr[4];
        {
            const int m  = lane & 15;
            const int kg = (lane >> 4) << 3;    // 0 | 8
            #pragma unroll
            for (int mi = 0; mi < 2; ++mi) {
                const __bf16* p = &As[buf][(wr * 32 + mi * 16 + m) * LDAS];
                af[mi] = concat8(*(const v8bf*)(p + kg),
                                 *(const v8bf*)(p + 16 + kg));
            }
            const int kh = (lane >> 4) << 3;    // 0 | 8
            #pragma unroll
            for (int ni = 0; ni < 4; ++ni) {
                const __bf16* p = &Bs[buf][(wc * 64 + ni * 16 + m) * LDBS + kh];
                bfr[ni] = concat8(*(const v8bf*)p, *(const v8bf*)(p + 16));
            }
        }
        #pragma unroll
        for (int mi = 0; mi < 2; ++mi)
            #pragma unroll
            for (int ni = 0; ni < 4; ++ni)
                acc[mi][ni] = __builtin_amdgcn_wmma_f32_16x16x32_bf16(
                    false, af[mi], false, bfr[ni], (short)0, acc[mi][ni],
                    false, false);
#if HAS_TDM
        if (tid < 32 && (i + 1) < nslab) __builtin_amdgcn_s_wait_tensorcnt(0);
#endif
        __syncthreads();
    }
    // ---- epilogue: one guard + one address per output row, 4 stores each ----
    // C/D layout: VGPR r -> rows r / r+8 (lane>=16); lanes 0..15 = column n.
    const int n       = lane & 15;
    const int mo      = (lane >> 4) << 3;
    const int colbase = col0 + wc * 64 + n;
    #pragma unroll
    for (int mi = 0; mi < 2; ++mi)
        #pragma unroll
        for (int r = 0; r < 8; ++r) {
            const int gr = row0 + wr * 32 + mi * 16 + mo + r;
            if (gr < N) {
                float* p = C + (size_t)gr * ldc + colbase;
                p[0]  = acc[mi][0][r];
                p[16] = acc[mi][1][r];
                p[32] = acc[mi][2][r];
                p[48] = acc[mi][3][r];
            }
        }
}

// --------------------- alpha_src / alpha_dst (wave/row dot) -----------------
__global__ __launch_bounds__(256)
void k_alpha(const float* __restrict__ XW, const float* __restrict__ a_src,
             const float* __restrict__ a_dst, float* __restrict__ as_out,
             float* __restrict__ ad_out, int N) {
    int w    = (blockIdx.x * 256 + threadIdx.x) >> 5;
    int lane = threadIdx.x & 31;
    if (w >= N) return;
    const float* row = XW + (size_t)w * DIM;
    float s0 = 0.f, s1 = 0.f;
    #pragma unroll
    for (int i = 0; i < 8; ++i) {
        int c = lane + i * 32;
        float v = row[c];
        s0 += v * a_src[c];
        s1 += v * a_dst[c];
    }
    #pragma unroll
    for (int o = 16; o > 0; o >>= 1) {
        s0 += __shfl_down(s0, o, 32);
        s1 += __shfl_down(s1, o, 32);
    }
    if (lane == 0) { as_out[w] = s0; ad_out[w] = s1; }
}

// -------------------------- edge softmax kernels ----------------------------
__global__ __launch_bounds__(256)
void k_edge_max(const long long* __restrict__ ei, const float* __restrict__ as,
                const float* __restrict__ ad, float* __restrict__ eval,
                float* __restrict__ m, int E, int N) {
    int e = blockIdx.x * 256 + threadIdx.x;
    if (e >= E + N) return;
    long long s, d;
    if (e < E) { s = ei[e]; d = ei[E + e]; } else { s = d = e - E; }
    float v = as[s] + ad[d];
    v = v > 0.f ? v : NEG_SLOPE * v;
    eval[e] = v;
    atomicMaxFloat(&m[d], v);
}

__global__ __launch_bounds__(256)
void k_edge_exp(const long long* __restrict__ ei, float* __restrict__ eval,
                const float* __restrict__ m, float* __restrict__ denom,
                int E, int N) {
    int e = blockIdx.x * 256 + threadIdx.x;
    if (e >= E + N) return;
    long long d = (e < E) ? ei[E + e] : (long long)(e - E);
    float v = __expf(eval[e] - m[d]);
    eval[e] = v;                        // in-place: e_val -> e_exp
    atomicAdd(&denom[d], v);
}

// wave per edge: out[dst] += alpha * xw[src]  (8 f32 / lane, coalesced)
__global__ __launch_bounds__(256)
void k_scatter(const long long* __restrict__ ei, const float* __restrict__ eexp,
               const float* __restrict__ denom, const float* __restrict__ XW,
               float* __restrict__ out, int E, int N) {
    int w    = (blockIdx.x * 256 + threadIdx.x) >> 5;
    int lane = threadIdx.x & 31;
    if (w >= E + N) return;
    long long s, d;
    if (w < E) { s = ei[w]; d = ei[E + w]; } else { s = d = w - E; }
    float alpha = eexp[w] / (denom[d] + 1e-16f);
    const float* srow = XW + (size_t)s * DIM;
    float*       drow = out + (size_t)d * DIM;
    #pragma unroll
    for (int i = 0; i < 8; ++i) {
        int c = lane + i * 32;
        atomicAdd(&drow[c], alpha * srow[c]);
    }
}

// xb = tanh(out + b) -> bf16 (4 elems / thread)
__global__ __launch_bounds__(256)
void k_finalize_xb(const float* __restrict__ out, const float* __restrict__ b,
                   __bf16* __restrict__ xb, int total4) {
    int i = blockIdx.x * 256 + threadIdx.x;
    if (i >= total4) return;
    float4 v = ((const float4*)out)[i];
    int c = (i * 4) & (DIM - 1);
    v4bf o;
    o[0] = (__bf16)tanhf(v.x + b[c]);
    o[1] = (__bf16)tanhf(v.y + b[c + 1]);
    o[2] = (__bf16)tanhf(v.z + b[c + 2]);
    o[3] = (__bf16)tanhf(v.w + b[c + 3]);
    ((v4bf*)xb)[i] = o;
}

// --------------------------- LSTM pointwise + output ------------------------
__global__ __launch_bounds__(256)
void k_lstm(const float* __restrict__ G, const float* __restrict__ c0,
            float* __restrict__ outp, int N) {
    int i = blockIdx.x * 256 + threadIdx.x;
    if (i >= N * DIM) return;
    int nrow = i >> 8, dcol = i & (DIM - 1);
    const float* g = G + (size_t)nrow * (4 * DIM);
    float iv = g[dcol], fv = g[DIM + dcol], gv = g[2 * DIM + dcol],
          ov = g[3 * DIM + dcol];
    float c1 = sigmoidf(fv) * c0[i] + sigmoidf(iv) * tanhf(gv);
    float h1 = sigmoidf(ov) * tanhf(c1);
    size_t ND = (size_t)N * DIM;
    outp[i]          = h1;   // h1
    outp[ND + i]     = h1;   // h1[None]
    outp[2 * ND + i] = c1;   // c1[None]
}

// ------------------------------- launcher -----------------------------------

extern "C" void kernel_launch(void* const* d_in, const int* in_sizes, int n_in,
                              void* d_out, int out_size, void* d_ws,
                              size_t ws_size, hipStream_t stream) {
    const float*     x     = (const float*)d_in[0];
    const long long* ei    = (const long long*)d_in[1];   // int64 [2, E]
    const float*     h0    = (const float*)d_in[2];
    const float*     c0    = (const float*)d_in[3];
    const float*     Wg    = (const float*)d_in[4];
    const float*     a_src = (const float*)d_in[5];
    const float*     a_dst = (const float*)d_in[6];
    const float*     b_gat = (const float*)d_in[7];
    const float*     Wih   = (const float*)d_in[8];
    const float*     Whh   = (const float*)d_in[9];
    float*           outp  = (float*)d_out;

    const int N  = in_sizes[0] / DIM;    // 100000
    const int E  = in_sizes[1] / 2;      // 800000
    const int EN = E + N;

    // ---- carve workspace (256B aligned) ----
    char*  base = (char*)d_ws;
    size_t off  = 0;
    auto carve = [&](size_t bytes) -> void* {
        void* p = base + off;
        off = (off + bytes + 255) & ~(size_t)255;
        return p;
    };
    float*  xw     = (float*)carve((size_t)N * DIM * 4);
    float*  gates  = (float*)carve((size_t)N * 4 * DIM * 4);
    float*  aggout = gates;  // alias: consumed (-> xb_bf) before gates written
    float*  asrc   = (float*)carve((size_t)N * 4);
    float*  adst   = (float*)carve((size_t)N * 4);
    float*  mseg   = (float*)carve((size_t)N * 4);
    float*  denom  = (float*)carve((size_t)N * 4);
    float*  eval   = (float*)carve((size_t)EN * 4);
    __bf16* Wg_sw  = (__bf16*)carve((size_t)DIM * DIM * 2);
    __bf16* WihT   = (__bf16*)carve((size_t)4 * DIM * DIM * 2);
    __bf16* WhhT   = (__bf16*)carve((size_t)4 * DIM * DIM * 2);
    __bf16* x_bf   = (__bf16*)carve((size_t)N * DIM * 2);
    __bf16* xb_bf  = (__bf16*)carve((size_t)N * DIM * 2);
    __bf16* h_bf   = (__bf16*)carve((size_t)N * DIM * 2);
    (void)ws_size; (void)n_in; (void)out_size;

    const int tilesM = (N + TM - 1) / TM;
    const int nd4    = (N * DIM) / 4;

    // 1) precision conversion: weights (swizzled), x, h
    k_cvt_weights<<<(4 * DIM * DIM + 255) / 256, 256, 0, stream>>>(
        Wg, Wih, Whh, Wg_sw, WihT, WhhT);
    k_cvt_f32_to_bf16_v4<<<(nd4 + 255) / 256, 256, 0, stream>>>(x, x_bf, nd4);
    k_cvt_f32_to_bf16_v4<<<(nd4 + 255) / 256, 256, 0, stream>>>(h0, h_bf, nd4);

    // 2) init segment-max / denom / aggregation buffer
    k_init<<<(N * DIM + 255) / 256, 256, 0, stream>>>(mseg, denom, aggout, N);

    // 3) xw = x @ W_gat (WMMA bf16, TDM-staged, double-buffered)
    k_gemm<<<dim3(tilesM, DIM / TN), 256, 0, stream>>>(
        x_bf, x_bf, Wg_sw, Wg_sw, xw, N, DIM, NKSLAB);

    // 4) alpha_src/alpha_dst
    k_alpha<<<(N * 32 + 255) / 256, 256, 0, stream>>>(xw, a_src, a_dst, asrc,
                                                      adst, N);
    // 5) edge softmax
    k_edge_max<<<(EN + 255) / 256, 256, 0, stream>>>(ei, asrc, adst, eval,
                                                     mseg, E, N);
    k_edge_exp<<<(EN + 255) / 256, 256, 0, stream>>>(ei, eval, mseg, denom,
                                                     E, N);
    // 6) attention-weighted scatter (wave per edge, xw L2-resident)
    k_scatter<<<((size_t)EN * 32 + 255) / 256, 256, 0, stream>>>(
        ei, eval, denom, xw, aggout, E, N);

    // 7) xb = tanh(agg + b) -> bf16
    k_finalize_xb<<<(nd4 + 255) / 256, 256, 0, stream>>>(aggout, b_gat, xb_bf,
                                                         nd4);
    // 8) gates = xb @ W_ihT + h @ W_hhT (fused two-phase WMMA)
    k_gemm<<<dim3(tilesM, (4 * DIM) / TN), 256, 0, stream>>>(
        xb_bf, h_bf, WihT, WhhT, gates, N, 4 * DIM, 2 * NKSLAB);

    // 9) LSTM pointwise -> d_out = [h1 | h1 | c1]
    k_lstm<<<(N * DIM + 255) / 256, 256, 0, stream>>>(gates, c0, outp, N);
}
